// GateProj_72722386256027
// MI455X (gfx1250) — matrix operand order repaired and checked by reading, not compiled, not run on hardware
//
#include <hip/hip_runtime.h>

// ---------------------------------------------------------------------------
// GateProj: y = silu( (qx @ dequant_int4(W)^T) * qxscale + bias )
//   M=4096, K=4096, N=14336, group G=128 (derived at launch; not hardcoded)
//
// Strategy (MI455X / gfx1250, wave32):
//   * bf16 WMMA (v_wmma_f32_16x16x32_bf16) -> compute-bound kernel rides the
//     matrix pipes. int4 weights are exact in bf16; group scale folded into
//     the dequantized weights (one scale per row per 128-wide K block).
//   * Workgroup = 256 threads = 8 waves. Output tile 128(M) x 128(N).
//     Each wave computes a 16(M) x 128(N) strip = 8 16x16 f32 accumulators.
//   * K loop in blocks of 128: stage A (fp32->bf16) and dequantized W into
//     LDS cooperatively, then 4 x (A-frag + 8 B-frags + 8 WMMA) per wave.
//   * global_prefetch_b8 on the next K block overlaps fetch with WMMA.
// ---------------------------------------------------------------------------

typedef __attribute__((ext_vector_type(16))) __bf16 v16bf;
typedef __attribute__((ext_vector_type(8)))  __bf16 v8bf;
typedef __attribute__((ext_vector_type(4)))  __bf16 v4bf;
typedef __attribute__((ext_vector_type(8)))  float  v8f;

#define BM  128
#define BN  128
#define BK  128
#define LDT (BK + 8)   // padded LDS row stride (elements) to dodge bank conflicts

__global__ __launch_bounds__(256)
void gateproj_w4a16_wmma(const float* __restrict__ qx,
                         const float* __restrict__ qxscale,
                         const int*   __restrict__ w4,      // [N, K/8] packed int4
                         const float* __restrict__ wscale,  // [N, K/G]
                         const float* __restrict__ bias,    // [N]
                         float*       __restrict__ out,     // [M, N]
                         int M, int K, int N, int G)
{
    __shared__ __bf16 sA[BM][LDT];   // activations, bf16
    __shared__ __bf16 sB[BN][LDT];   // dequantized weights, bf16 (scale folded)

    const int tid  = threadIdx.x;
    const int wave = tid >> 5;       // 0..7
    const int lane = tid & 31;
    const int hh   = lane >> 4;      // half-wave select
    const int lr   = lane & 15;

    const int bn = blockIdx.x * BN;
    const int bm = blockIdx.y * BM;

    const int Kd8 = K >> 3;          // int32 words per weight row
    const int KdG = K / G;           // scale groups per row

    v8f acc[8];
    #pragma unroll
    for (int t = 0; t < 8; ++t)
        #pragma unroll
        for (int j = 0; j < 8; ++j) acc[t][j] = 0.0f;

    const int kblocks = K / BK;
    for (int kb = 0; kb < kblocks; ++kb) {
        const int k0 = kb * BK;

        // ---- stage A tile: 128 x 128 fp32 -> bf16 in LDS -------------------
        // 16384 floats / 256 threads = 16 float4 per thread, fully coalesced.
        #pragma unroll 4
        for (int i = 0; i < 16; ++i) {
            const int e  = tid + i * 256;     // float4 slot in tile
            const int r  = e >> 5;            // 0..127
            const int c4 = e & 31;            // 0..31 (float4 column)
            const float4 v =
                *(const float4*)(qx + (size_t)(bm + r) * K + k0 + c4 * 4);
            v4bf pk;
            pk[0] = (__bf16)v.x; pk[1] = (__bf16)v.y;
            pk[2] = (__bf16)v.z; pk[3] = (__bf16)v.w;
            *(v4bf*)&sA[r][c4 * 4] = pk;      // ds_store_b64
        }

        // ---- stage W tile: 128 rows x 16 dwords int4 -> bf16*scale in LDS --
        // 2048 dwords / 256 threads = 2 x int4(16B) loads per thread.
        #pragma unroll
        for (int i = 0; i < 2; ++i) {
            const int e   = tid + i * 256;    // 0..511
            const int r   = e >> 2;           // 0..127 (N row in tile)
            const int seg = e & 3;            // 32-K-value segment
            const int4 w = *(const int4*)(w4 + (size_t)(bn + r) * Kd8
                                             + (k0 >> 3) + seg * 4);
            const float s = wscale[(size_t)(bn + r) * KdG + (k0 + seg * 32) / G];
            int ws[4] = {w.x, w.y, w.z, w.w};
            __bf16* dst = &sB[r][seg * 32];
            #pragma unroll
            for (int d = 0; d < 4; ++d) {
                const int wv = ws[d];
                v8bf pk;
                #pragma unroll
                for (int j = 0; j < 8; ++j) {
                    int q = (wv >> (4 * j)) & 0xF;
                    q = (q ^ 8) - 8;                   // signed int4
                    pk[j] = (__bf16)((float)q * s);    // fold group scale
                }
                *(v8bf*)(dst + d * 8) = pk;            // ds_store_b128
            }
        }

        // ---- prefetch next K block toward cache while we compute ----------
        if (kb + 1 < kblocks) {
            const int pr = tid >> 1;
            __builtin_prefetch(qx + (size_t)(bm + pr) * K + (k0 + BK)
                                  + (tid & 1) * 64, 0, 0);
            __builtin_prefetch(w4 + (size_t)(bn + pr) * Kd8 + ((k0 + BK) >> 3),
                               0, 0);
        }

        __syncthreads();

        // ---- WMMA over this K block ---------------------------------------
        // A frag (16x32 bf16): lane<16 holds K 0..15, lane>=16 holds
        // K 16..31 of row M=lr. B frag (32x16): half-wave selects
        // K 0..15 / 16..31, N = lr. All 16B LDS reads -> ds_load_b128.
        const int mrow = wave * 16 + lr;
        #pragma unroll
        for (int kk = 0; kk < 4; ++kk) {
            const int kb32 = kk * 32;
            v16bf a;
            {
                const v8bf lo = *(const v8bf*)&sA[mrow][kb32 + hh * 16];
                const v8bf hi = *(const v8bf*)&sA[mrow][kb32 + hh * 16 + 8];
                #pragma unroll
                for (int j = 0; j < 8; ++j) { a[j] = lo[j]; a[j + 8] = hi[j]; }
            }
            #pragma unroll
            for (int t = 0; t < 8; ++t) {
                v16bf b;
                const v8bf b0 = *(const v8bf*)&sB[t * 16 + lr][kb32 + hh * 16];
                const v8bf b1 = *(const v8bf*)&sB[t * 16 + lr][kb32 + hh * 16 + 8];
                #pragma unroll
                for (int j = 0; j < 8; ++j) { b[j] = b0[j]; b[j + 8] = b1[j]; }
                acc[t] = __builtin_amdgcn_wmma_f32_16x16x32_bf16(
                    false, a, false, b, (short)0, acc[t], false, false);
            }
        }
        __syncthreads();
    }

    // ---- epilogue: qxscale * acc + bias, SiLU, store -----------------------
    // C/D layout: VGPR r, lane<16 -> M=r, lane>=16 -> M=r+8; N = lane%16.
    float qs[8];
    #pragma unroll
    for (int r = 0; r < 8; ++r)
        qs[r] = qxscale[bm + wave * 16 + hh * 8 + r];

    #pragma unroll
    for (int t = 0; t < 8; ++t) {
        const int gcol = bn + t * 16 + lr;
        const float bv = bias[gcol];
        #pragma unroll
        for (int r = 0; r < 8; ++r) {
            const int grow = bm + wave * 16 + hh * 8 + r;
            float y = acc[t][r] * qs[r] + bv;
            const float sig = 1.0f / (1.0f + __expf(-y));
            out[(size_t)grow * N + gcol] = y * sig;
        }
    }
}

extern "C" void kernel_launch(void* const* d_in, const int* in_sizes, int n_in,
                              void* d_out, int out_size, void* d_ws, size_t ws_size,
                              hipStream_t stream)
{
    const float* qx      = (const float*)d_in[0];
    const float* qxscale = (const float*)d_in[1];
    const int*   w4      = (const int*)d_in[2];
    const float* wscale  = (const float*)d_in[3];
    const float* bias    = (const float*)d_in[4];
    float*       out     = (float*)d_out;

    const int M = in_sizes[1];                         // qxscale is [M,1]
    const int K = (int)((long long)in_sizes[0] / M);   // qx is [M,K]
    const int N = in_sizes[4];                         // bias is [N]
    const int G = (int)(((long long)N * K) / in_sizes[3]); // wscale is [N,K/G]

    dim3 grid(N / BN, M / BM);   // 112 x 32 for the reference shapes
    gateproj_w4a16_wmma<<<grid, 256, 0, stream>>>(
        qx, qxscale, w4, wscale, bias, out, M, K, N, G);
}
